// GraphConvLayer_56513179680870
// MI455X (gfx1250) — matrix operand order repaired and checked by reading, not compiled, not run on hardware
//
#include <hip/hip_runtime.h>

typedef __attribute__((ext_vector_type(16))) _Float16 v16h;
typedef __attribute__((ext_vector_type(8)))  _Float16 v8h;
typedef __attribute__((ext_vector_type(8)))  float    v8f;

#define DD 128   // input feature dim
#define HH 128   // hidden dim

union V16U { v16h v; struct { v8h lo; v8h hi; } s; };

// A-operand (16x32 f16) chunk load. Per ISA table:
// lanes 0-15:  K 0-7 in VGPR0-3, K 16-23 in VGPR4-7
// lanes 16-31: K 8-15,           K 24-31
__device__ __forceinline__ v16h load_a_chunk(const _Float16* base, int sel) {
  V16U u;
  u.s.lo = *(const v8h*)(base + sel * 8);
  u.s.hi = *(const v8h*)(base + 16 + sel * 8);
  return u.v;
}

// B-operand (32x16 f16) chunk load, column-major weights (col stride = K).
// lanes 0-15 hold K 0-7, 16-23 for col=lane; lanes 16-31 hold K 8-15, 24-31.
__device__ __forceinline__ v16h load_b_chunk(const _Float16* base, int sel) {
  V16U u;
  u.s.lo = *(const v8h*)(base + sel * 8);
  u.s.hi = *(const v8h*)(base + 16 + sel * 8);
  return u.v;
}

__device__ __forceinline__ v8f wmma16(v16h a, v16h b, v8f c) {
  return __builtin_amdgcn_wmma_f32_16x16x32_f16(
      /*neg_a=*/false, a, /*neg_b=*/false, b,
      /*c_mod=*/(short)0, c, /*reuse_a=*/false, /*reuse_b=*/false);
}

// Branchless exact-ish gelu: 0.5*x*(1+erf(x/sqrt(2))) with Abramowitz-Stegun
// 7.1.26 erf (|err| < 1.5e-7 on erf). One v_rcp_f32 + one v_exp_f32, no branches.
__device__ __forceinline__ float gelu_exact(float x) {
  float z = x * 0.70710678118654752f;
  float a = __builtin_fabsf(z);
  float t = __builtin_amdgcn_rcpf(__builtin_fmaf(0.3275911f, a, 1.0f));
  float p = __builtin_fmaf(1.061405429f, t, -1.453152027f);
  p = __builtin_fmaf(p, t, 1.421413741f);
  p = __builtin_fmaf(p, t, -0.284496736f);
  p = __builtin_fmaf(p, t, 0.254829592f);
  p = p * t;
  float e = __builtin_amdgcn_exp2f(-a * a * 1.4426950408889634f);  // exp(-a^2)
  float erfa = __builtin_fmaf(-p, e, 1.0f);                        // erf(|z|)
  float erfz = __builtin_copysignf(erfa, z);
  return 0.5f * x * (1.0f + erfz);
}

// ---------------- BN fold: W'[k][n] = s[k]*W[k][n] (stored col-major f16),
// bias'[n] = bias[n] + sum_k t[k]*W[k][n],  s=g*rsqrt(v+eps), t=b-m*s --------
__global__ void bn_fold_kernel(const float* __restrict__ g, const float* __restrict__ b,
                               const float* __restrict__ m, const float* __restrict__ v,
                               const float* __restrict__ W, const float* __restrict__ bias,
                               _Float16* __restrict__ Wc, float* __restrict__ bout, int K) {
  int col = threadIdx.x;
  if (col >= HH) return;
  float acc = bias[col];
  for (int k = 0; k < K; ++k) {
    float s = g[k] * rsqrtf(v[k] + 1e-3f);
    float t = b[k] - m[k] * s;
    float w = W[(size_t)k * HH + col];
    Wc[(size_t)col * K + k] = (_Float16)(s * w);
    acc += t * w;
  }
  bout[col] = acc;
}

// ---------------- f32 -> f16 convert ----------------
__global__ void cvt_f16_kernel(const float* __restrict__ in, _Float16* __restrict__ out, int n) {
  int i = blockIdx.x * blockDim.x + threadIdx.x;
  if (i < n) out[i] = (_Float16)in[i];
}

// ---------------- per-node incoming-edge counts ----------------
__global__ void count_kernel(const int* __restrict__ edges, float* __restrict__ cnt, int E_) {
  int e = blockIdx.x * blockDim.x + threadIdx.x;
  if (e < E_) {
    int ni = edges[e];  // edges[0][e] = destination node
    __hip_atomic_fetch_add(&cnt[ni], 1.0f, __ATOMIC_RELAXED, __HIP_MEMORY_SCOPE_AGENT);
  }
}

// ---------------- fused prep FFN (2 layers) + weighted scatter-add ----------
// 8 waves / block, 16 edges / wave -> 128 edges / block.
// Dynamic LDS: lW0 (32KB) | lW1 (32KB) | per-wave X1 bounce (32KB) = 96KB.
__global__ __launch_bounds__(256) void prep_scatter_kernel(
    const _Float16* __restrict__ nodeF16,
    const int* __restrict__ edges,        // [2,E] flat
    const float* __restrict__ ew,
    const _Float16* __restrict__ W0c, const float* __restrict__ b0,
    const _Float16* __restrict__ W1c, const float* __restrict__ b1,
    float* __restrict__ ssum, int E_) {
  extern __shared__ __align__(16) char dynsmem[];
  _Float16* lW0 = (_Float16*)dynsmem;                 // 128*128
  _Float16* lW1 = lW0 + DD * HH;                      // 128*128
  _Float16* lX1 = lW1 + HH * HH;                      // 8 * 16*128

  const int tid  = threadIdx.x;
  const int wave = tid >> 5;
  const int lane = tid & 31;
  const int l15  = lane & 15;
  const int sel  = lane >> 4;
  const int e0   = blockIdx.x * 128 + wave * 16;

  // Cooperative weight staging (shared by all 8 waves)
  for (int i = tid * 8; i < DD * HH; i += 256 * 8) {
    *(v8h*)&lW0[i] = *(const v8h*)&W0c[i];
    *(v8h*)&lW1[i] = *(const v8h*)&W1c[i];
  }
  __syncthreads();

  int erow = e0 + l15; if (erow > E_ - 1) erow = E_ - 1;
  const int nbr = edges[E_ + erow];       // edges[1][erow] = neighbour
  const _Float16* abase = nodeF16 + (size_t)nbr * DD;

  v16h A[4];
  #pragma unroll
  for (int kc = 0; kc < 4; ++kc) A[kc] = load_a_chunk(abase + kc * 32, sel);

  _Float16* myX1 = lX1 + wave * (16 * HH);

  // ---- layer 0: X1 = gelu(X @ W0' + b0') ----
  #pragma unroll
  for (int t = 0; t < 8; ++t) {
    const int col = t * 16 + l15;
    float bc = b0[col];
    v8f acc;
    #pragma unroll
    for (int i = 0; i < 8; ++i) acc[i] = bc;
    v16h B[4];
    #pragma unroll
    for (int kc = 0; kc < 4; ++kc) B[kc] = load_b_chunk(lW0 + col * DD + kc * 32, sel);
    #pragma unroll
    for (int kc = 0; kc < 4; ++kc) acc = wmma16(A[kc], B[kc], acc);
    #pragma unroll
    for (int r = 0; r < 8; ++r) {
      int row = r + 8 * sel;              // C-layout: VGPR r -> row r (+8 for hi lanes)
      myX1[row * HH + col] = (_Float16)gelu_exact(acc[r]);
    }
  }
  __syncthreads();

  // ---- layer 1: M = gelu(X1 @ W1' + b1') * ew, scatter-add ----
  const _Float16* a1base = myX1 + l15 * HH;
  #pragma unroll
  for (int kc = 0; kc < 4; ++kc) A[kc] = load_a_chunk(a1base + kc * 32, sel);

  #pragma unroll
  for (int t = 0; t < 8; ++t) {
    const int col = t * 16 + l15;
    float bc = b1[col];
    v8f acc;
    #pragma unroll
    for (int i = 0; i < 8; ++i) acc[i] = bc;
    v16h B[4];
    #pragma unroll
    for (int kc = 0; kc < 4; ++kc) B[kc] = load_b_chunk(lW1 + col * HH + kc * 32, sel);
    #pragma unroll
    for (int kc = 0; kc < 4; ++kc) acc = wmma16(A[kc], B[kc], acc);
    #pragma unroll
    for (int r = 0; r < 8; ++r) {
      int row = r + 8 * sel;
      int e = e0 + row;
      if (e < E_) {
        float val = gelu_exact(acc[r]) * ew[e];
        int ni = edges[e];
        __hip_atomic_fetch_add(&ssum[(size_t)ni * HH + col], val,
                               __ATOMIC_RELAXED, __HIP_MEMORY_SCOPE_AGENT);
      }
    }
  }
}

// ---------------- segment mean -> f16 ----------------
__global__ void agg_kernel(const float* __restrict__ ssum, const float* __restrict__ cnt,
                           _Float16* __restrict__ aggF16, int total) {
  int i = blockIdx.x * blockDim.x + threadIdx.x;
  if (i < total) {
    float c = cnt[i >> 7];
    aggF16[i] = (_Float16)(c > 0.0f ? ssum[i] / c : 0.0f);
  }
}

// ---------------- fused update FFN: concat(node, agg) -> 2 layers -> out ----
// Dynamic LDS: lW0 (64KB) | lW1 (32KB) | per-wave X1 bounce (32KB) = 128KB.
__global__ __launch_bounds__(256) void update_kernel(
    const _Float16* __restrict__ nodeF16, const _Float16* __restrict__ aggF16,
    const _Float16* __restrict__ W0c, const float* __restrict__ b0,  // K=256 col-major
    const _Float16* __restrict__ W1c, const float* __restrict__ b1,  // K=128 col-major
    float* __restrict__ out, int N_) {
  extern __shared__ __align__(16) char dynsmem[];
  _Float16* lW0 = (_Float16*)dynsmem;                 // 256*128
  _Float16* lW1 = lW0 + (DD + HH) * HH;               // 128*128
  _Float16* lX1 = lW1 + HH * HH;                      // 8 * 16*128

  const int tid  = threadIdx.x;
  const int wave = tid >> 5;
  const int lane = tid & 31;
  const int l15  = lane & 15;
  const int sel  = lane >> 4;
  const int n0   = blockIdx.x * 128 + wave * 16;

  for (int i = tid * 8; i < (DD + HH) * HH; i += 256 * 8) *(v8h*)&lW0[i] = *(const v8h*)&W0c[i];
  for (int i = tid * 8; i < HH * HH;        i += 256 * 8) *(v8h*)&lW1[i] = *(const v8h*)&W1c[i];
  __syncthreads();

  int nrow = n0 + l15; if (nrow > N_ - 1) nrow = N_ - 1;
  const _Float16* nb = nodeF16 + (size_t)nrow * DD;
  const _Float16* ab = aggF16 + (size_t)nrow * HH;

  v16h A[8];  // K=256: chunks 0-3 from node features, 4-7 from aggregate
  #pragma unroll
  for (int kc = 0; kc < 4; ++kc) A[kc]     = load_a_chunk(nb + kc * 32, sel);
  #pragma unroll
  for (int kc = 0; kc < 4; ++kc) A[4 + kc] = load_a_chunk(ab + kc * 32, sel);

  _Float16* myX1 = lX1 + wave * (16 * HH);

  // ---- layer 0 (K=256) ----
  #pragma unroll
  for (int t = 0; t < 8; ++t) {
    const int col = t * 16 + l15;
    float bc = b0[col];
    v8f acc;
    #pragma unroll
    for (int i = 0; i < 8; ++i) acc[i] = bc;
    #pragma unroll
    for (int half = 0; half < 2; ++half) {
      v16h B[4];
      #pragma unroll
      for (int kc = 0; kc < 4; ++kc)
        B[kc] = load_b_chunk(lW0 + col * 256 + (half * 4 + kc) * 32, sel);
      #pragma unroll
      for (int kc = 0; kc < 4; ++kc) acc = wmma16(A[half * 4 + kc], B[kc], acc);
    }
    #pragma unroll
    for (int r = 0; r < 8; ++r) {
      int row = r + 8 * sel;
      myX1[row * HH + col] = (_Float16)gelu_exact(acc[r]);
    }
  }
  __syncthreads();

  // ---- layer 1 (K=128) ----
  const _Float16* a1base = myX1 + l15 * HH;
  #pragma unroll
  for (int kc = 0; kc < 4; ++kc) A[kc] = load_a_chunk(a1base + kc * 32, sel);

  #pragma unroll
  for (int t = 0; t < 8; ++t) {
    const int col = t * 16 + l15;
    float bc = b1[col];
    v8f acc;
    #pragma unroll
    for (int i = 0; i < 8; ++i) acc[i] = bc;
    v16h B[4];
    #pragma unroll
    for (int kc = 0; kc < 4; ++kc) B[kc] = load_b_chunk(lW1 + col * HH + kc * 32, sel);
    #pragma unroll
    for (int kc = 0; kc < 4; ++kc) acc = wmma16(A[kc], B[kc], acc);
    #pragma unroll
    for (int r = 0; r < 8; ++r) {
      int row = r + 8 * sel;
      int n = n0 + row;
      if (n < N_) out[(size_t)n * HH + col] = gelu_exact(acc[r]);
    }
  }
}

extern "C" void kernel_launch(void* const* d_in, const int* in_sizes, int n_in,
                              void* d_out, int out_size, void* d_ws, size_t ws_size,
                              hipStream_t stream) {
  const float* node_repr = (const float*)d_in[0];
  const int*   edges     = (const int*)d_in[1];
  const float* ew        = (const float*)d_in[2];
  // prep layer0: 3..8, prep layer1: 9..14, upd layer0: 15..20, upd layer1: 21..26
  const float* pg0 = (const float*)d_in[3],  *pb0_ = (const float*)d_in[4];
  const float* pm0 = (const float*)d_in[5],  *pv0  = (const float*)d_in[6];
  const float* pW0 = (const float*)d_in[7],  *pbi0 = (const float*)d_in[8];
  const float* pg1 = (const float*)d_in[9],  *pb1_ = (const float*)d_in[10];
  const float* pm1 = (const float*)d_in[11], *pv1  = (const float*)d_in[12];
  const float* pW1 = (const float*)d_in[13], *pbi1 = (const float*)d_in[14];
  const float* ug0 = (const float*)d_in[15], *ub0_ = (const float*)d_in[16];
  const float* um0 = (const float*)d_in[17], *uv0  = (const float*)d_in[18];
  const float* uW0 = (const float*)d_in[19], *ubi0 = (const float*)d_in[20];
  const float* ug1 = (const float*)d_in[21], *ub1_ = (const float*)d_in[22];
  const float* um1 = (const float*)d_in[23], *uv1  = (const float*)d_in[24];
  const float* uW1 = (const float*)d_in[25], *ubi1 = (const float*)d_in[26];

  const int N_ = in_sizes[0] / DD;
  const int E_ = in_sizes[1] / 2;

  // ---- workspace carve (256B aligned) ----
  char* w = (char*)d_ws;
  auto carve = [&](size_t bytes) -> void* {
    void* p = (void*)w;
    w += (bytes + 255) & ~(size_t)255;
    return p;
  };
  float*     ssum    = (float*)    carve((size_t)N_ * HH * sizeof(float));
  float*     cnt     = (float*)    carve((size_t)N_ * sizeof(float));
  _Float16*  nodeF16 = (_Float16*) carve((size_t)N_ * DD * sizeof(_Float16));
  _Float16*  aggF16  = (_Float16*) carve((size_t)N_ * HH * sizeof(_Float16));
  _Float16*  pW0c    = (_Float16*) carve((size_t)DD * HH * sizeof(_Float16));
  _Float16*  pW1c    = (_Float16*) carve((size_t)HH * HH * sizeof(_Float16));
  _Float16*  uW0c    = (_Float16*) carve((size_t)(DD + HH) * HH * sizeof(_Float16));
  _Float16*  uW1c    = (_Float16*) carve((size_t)HH * HH * sizeof(_Float16));
  float*     pb0f    = (float*)    carve(HH * sizeof(float));
  float*     pb1f    = (float*)    carve(HH * sizeof(float));
  float*     ub0f    = (float*)    carve(HH * sizeof(float));
  float*     ub1f    = (float*)    carve(HH * sizeof(float));

  // 1) fold BN into weights (col-major f16) + biases
  bn_fold_kernel<<<1, 128, 0, stream>>>(pg0, pb0_, pm0, pv0, pW0, pbi0, pW0c, pb0f, DD);
  bn_fold_kernel<<<1, 128, 0, stream>>>(pg1, pb1_, pm1, pv1, pW1, pbi1, pW1c, pb1f, HH);
  bn_fold_kernel<<<1, 128, 0, stream>>>(ug0, ub0_, um0, uv0, uW0, ubi0, uW0c, ub0f, DD + HH);
  bn_fold_kernel<<<1, 128, 0, stream>>>(ug1, ub1_, um1, uv1, uW1, ubi1, uW1c, ub1f, HH);

  // 2) pre-convert node features to f16
  {
    int total = N_ * DD;
    cvt_f16_kernel<<<(total + 255) / 256, 256, 0, stream>>>(node_repr, nodeF16, total);
  }

  // 3) zero accumulators
  hipMemsetAsync(ssum, 0, (size_t)N_ * HH * sizeof(float), stream);
  hipMemsetAsync(cnt, 0, (size_t)N_ * sizeof(float), stream);

  // 4) per-node incoming edge counts
  count_kernel<<<(E_ + 255) / 256, 256, 0, stream>>>(edges, cnt, E_);

  // 5) fused per-edge FFN + weighted scatter-add (96KB dynamic LDS)
  {
    size_t ldsB = (size_t)(DD * HH + HH * HH + 8 * 16 * HH) * sizeof(_Float16);
    prep_scatter_kernel<<<(E_ + 127) / 128, 256, ldsB, stream>>>(
        nodeF16, edges, ew, pW0c, pb0f, pW1c, pb1f, ssum, E_);
  }

  // 6) segment mean -> f16
  {
    int total = N_ * HH;
    agg_kernel<<<(total + 255) / 256, 256, 0, stream>>>(ssum, cnt, aggF16, total);
  }

  // 7) fused update FFN -> output (128KB dynamic LDS)
  {
    size_t ldsB = (size_t)((DD + HH) * HH + HH * HH + 8 * 16 * HH) * sizeof(_Float16);
    update_kernel<<<(N_ + 127) / 128, 256, ldsB, stream>>>(
        nodeF16, aggF16, uW0c, ub0f, uW1c, ub1f, (float*)d_out, N_);
  }
}
